// EPCLDetection_20461224198850
// MI455X (gfx1250) — compile-verified, matrix operand order checked by reading, not run on hardware
//
#include <hip/hip_runtime.h>

// ---------------------------------------------------------------------------
// Types for CDNA5 WMMA
// ---------------------------------------------------------------------------
typedef _Float16 v16h __attribute__((ext_vector_type(16)));
typedef _Float16 v8h  __attribute__((ext_vector_type(8)));
typedef float    v8f  __attribute__((ext_vector_type(8)));

#define N0C 65536
#define N1C 16384
#define N2C 4096
#define N3C 1024
#define MC  512
#define KKC 27
#define EPSC 1e-5f

// ---------------------------------------------------------------------------
// Weight pre-pack: f32 [Kk,Cin,Cout] -> f16 WMMA-fragment order
//   wp[(((k*CC + cc)*2 + hi)*Cout + col)*16 + e]
//     = w[(k*Cin + cc*32 + hi*8 + (e<8 ? e : e+8))*Cout + col]
// so a lane's 16 B-operand halves are one contiguous, 32B-aligned chunk.
// ---------------------------------------------------------------------------
__global__ void pack_w(const float* __restrict__ w, _Float16* __restrict__ wp,
                       int Cin, int Cout, int Kk)
{
    size_t total = (size_t)Kk * Cin * Cout;
    size_t i = (size_t)blockIdx.x * blockDim.x + threadIdx.x;
    if (i >= total) return;
    const int CC = Cin >> 5;
    int e = (int)(i & 15);
    size_t r = i >> 4;
    int col = (int)(r % Cout); r /= Cout;
    int hi  = (int)(r & 1);    r >>= 1;
    int cc  = (int)(r % CC);
    int k   = (int)(r / CC);
    int K = cc * 32 + hi * 8 + (e < 8 ? e : e + 8);
    wp[i] = (_Float16)w[((size_t)k * Cin + K) * Cout + col];
}

// ---------------------------------------------------------------------------
// Generic sparse-conv / GEMM kernel on v_wmma_f32_16x16x32_f16.
// out[n,d] = sum_k sum_c xh[nbr[n,k], c] * wp(k,c,d)
// One wave computes a 16x32 (rows x out-channels) tile: the A fragment is
// reused by two independent WMMAs (keeps the matrix pipe busy, halves gather
// traffic).  B fragments are single v16h (32B) coalesced loads from the
// packed-f16 weights.  Cin % 32 == 0, Cout % 64 == 0, Nout % 16 == 0.
// ---------------------------------------------------------------------------
__global__ void __launch_bounds__(128) sconv_wmma(
    const _Float16* __restrict__ xh, const int* __restrict__ nbr,
    const _Float16* __restrict__ wp, float* __restrict__ out,
    int Nout, int Cin, int Cout, int Kk)
{
    const int lane = threadIdx.x;      // 0..31 (wave32)
    const int wave = threadIdx.y;
    const int lr   = lane & 15;
    const int hi   = lane >> 4;        // 0 or 1
    const int kofs = hi * 8;
    const int m0   = blockIdx.x * 16;
    const int n0   = (blockIdx.y * blockDim.y + wave) * 32;
    if (n0 >= Cout) return;            // wave-uniform: EXEC stays all-ones

    const int row  = m0 + lr;
    const int col0 = n0 + lr;
    const int col1 = n0 + 16 + lr;
    const int CC   = Cin >> 5;

    v8f acc0 = {}, acc1 = {};
    for (int k = 0; k < Kk; ++k) {
        const int src = nbr[(size_t)row * Kk + k];
        const _Float16* xr = xh + (size_t)src * Cin;
        if (k + 1 < Kk) {              // hide gather latency behind this tap
            const int nsrc = nbr[(size_t)row * Kk + k + 1];
            __builtin_prefetch(xh + (size_t)nsrc * Cin, 0, 0);
        }
        const _Float16* wk = wp + (size_t)k * CC * 2 * Cout * 16;
        for (int cc = 0; cc < CC; ++cc) {
            const int c0 = cc * 32;
            // A operand: two contiguous 16B chunks per lane
            v8h lo = *(const v8h*)(xr + c0 + kofs);
            v8h hv = *(const v8h*)(xr + c0 + kofs + 16);
            v16h a;
#pragma unroll
            for (int e = 0; e < 8; ++e) { a[e] = lo[e]; a[e + 8] = hv[e]; }
            // B operands: one 32B vector load per column tile, coalesced
            const _Float16* fb = wk + (((size_t)cc * 2 + hi) * Cout) * 16;
            v16h b0 = *(const v16h*)(fb + (size_t)col0 * 16);
            v16h b1 = *(const v16h*)(fb + (size_t)col1 * 16);
            acc0 = __builtin_amdgcn_wmma_f32_16x16x32_f16(
                false, a, false, b0, (short)0, acc0, false, false);
            acc1 = __builtin_amdgcn_wmma_f32_16x16x32_f16(
                false, a, false, b1, (short)0, acc1, false, false);
        }
    }
    // D layout: VGPR v holds row m0 + v + 8*hi
#pragma unroll
    for (int v = 0; v < 8; ++v) {
        out[(size_t)(m0 + hi * 8 + v) * Cout + col0] = acc0[v];
        out[(size_t)(m0 + hi * 8 + v) * Cout + col1] = acc1[v];
    }
}

// ---------------------------------------------------------------------------
// Small helpers
// ---------------------------------------------------------------------------
__global__ void zero_k(float* p, int n) {
    for (int i = threadIdx.x; i < n; i += blockDim.x) p[i] = 0.f;
}

__global__ void iota_k(int* p, int n) {
    int i = blockIdx.x * blockDim.x + threadIdx.x;
    if (i < n) p[i] = i;
}

// out[n,c] = sum_{j<3} a[n,j] * w[j,c]
__global__ void matmul3(const float* __restrict__ a, const float* __restrict__ w,
                        float* __restrict__ out, int N, int C)
{
    size_t i = (size_t)blockIdx.x * blockDim.x + threadIdx.x;
    if (i >= (size_t)N * C) return;
    int n = (int)(i / C), c = (int)(i % C);
    out[i] = a[n * 3] * w[c] + a[n * 3 + 1] * w[C + c] + a[n * 3 + 2] * w[2 * C + c];
}

// BatchNorm statistics: stats[0..C) = sum, stats[C..2C) = sum of squares
__global__ void bn_stats(const float* __restrict__ x, float* __restrict__ stats,
                         int N, int C)
{
    __shared__ float ss[1024];
    for (int i = threadIdx.x; i < 2 * C; i += blockDim.x) ss[i] = 0.f;
    __syncthreads();
    size_t total = (size_t)N * C;
    for (size_t i = (size_t)blockIdx.x * blockDim.x + threadIdx.x; i < total;
         i += (size_t)gridDim.x * blockDim.x) {
        float v = x[i];
        int c = (int)(i % C);
        atomicAdd(&ss[c], v);
        atomicAdd(&ss[C + c], v * v);
    }
    __syncthreads();
    for (int c = threadIdx.x; c < 2 * C; c += blockDim.x)
        atomicAdd(&stats[c], ss[c]);
}

__global__ void bn_finalize(const float* __restrict__ stats,
                            const float* __restrict__ g, const float* __restrict__ b,
                            float* __restrict__ scsh, int N, int C)
{
    int c = blockIdx.x * blockDim.x + threadIdx.x;
    if (c >= C) return;
    float mean = stats[c] / (float)N;
    float var  = stats[C + c] / (float)N - mean * mean;
    float sc   = g[c] * rsqrtf(var + EPSC);
    scsh[c]     = sc;
    scsh[C + c] = b[c] - mean * sc;
}

// v = raw*scale + shift + res1; if relu v=max(v,0); v += res2; write f32/f16
__global__ void apply_ew(const float* __restrict__ raw,
                         const float* __restrict__ scale, const float* __restrict__ shift,
                         const float* __restrict__ res1, const float* __restrict__ res2,
                         float* __restrict__ outf, _Float16* __restrict__ outh,
                         int N, int C, int relu)
{
    size_t i = (size_t)blockIdx.x * blockDim.x + threadIdx.x;
    if (i >= (size_t)N * C) return;
    int c = (int)(i % C);
    float v = raw[i];
    if (scale) v *= scale[c];
    if (shift) v += shift[c];
    if (res1)  v += res1[i];
    if (relu)  v = fmaxf(v, 0.f);
    if (res2)  v += res2[i];
    if (outf) outf[i] = v;
    if (outh) outh[i] = (_Float16)v;
}

// LayerNorm over rows of C==256, one block per row, writes f16
__global__ void ln_rows(const float* __restrict__ x, const float* __restrict__ g,
                        const float* __restrict__ b, _Float16* __restrict__ outh, int C)
{
    __shared__ float red[256];
    int m = blockIdx.x, t = threadIdx.x;
    float v = x[(size_t)m * C + t];
    red[t] = v; __syncthreads();
    for (int s = 128; s > 0; s >>= 1) { if (t < s) red[t] += red[t + s]; __syncthreads(); }
    float mean = red[0] / (float)C;
    __syncthreads();
    float d = v - mean;
    red[t] = d * d; __syncthreads();
    for (int s = 128; s > 0; s >>= 1) { if (t < s) red[t] += red[t + s]; __syncthreads(); }
    float var = red[0] / (float)C;
    outh[(size_t)m * C + t] = (_Float16)(d * rsqrtf(var + EPSC) * g[t] + b[t]);
}

// softmax attention, one block (128 threads) per (token m, head h)
__global__ void attn_kernel(const float* __restrict__ q, const float* __restrict__ k,
                            const float* __restrict__ vv, _Float16* __restrict__ outh,
                            int Mn)
{
    __shared__ float p[512];
    __shared__ float red[128];
    const int H = 8, dh = 32, D = 256;
    int m = blockIdx.x / H, h = blockIdx.x % H, t = threadIdx.x;
    const float* qm = q + (size_t)m * D + h * dh;
    const float scale = 0.17677669529663687f; // 1/sqrt(32)
    float lmax = -1e30f;
    for (int n = t; n < Mn; n += 128) {
        const float* kn = k + (size_t)n * D + h * dh;
        float dot = 0.f;
#pragma unroll
        for (int d = 0; d < dh; ++d) dot += qm[d] * kn[d];
        dot *= scale;
        p[n] = dot;
        lmax = fmaxf(lmax, dot);
    }
    red[t] = lmax; __syncthreads();
    for (int s = 64; s > 0; s >>= 1) { if (t < s) red[t] = fmaxf(red[t], red[t + s]); __syncthreads(); }
    float mx = red[0]; __syncthreads();
    float lsum = 0.f;
    for (int n = t; n < Mn; n += 128) { float e = __expf(p[n] - mx); p[n] = e; lsum += e; }
    red[t] = lsum; __syncthreads();
    for (int s = 64; s > 0; s >>= 1) { if (t < s) red[t] += red[t + s]; __syncthreads(); }
    float inv = 1.f / red[0];
    __syncthreads();
    if (t < dh) {
        float acc = 0.f;
        for (int n = 0; n < Mn; ++n) acc += p[n] * vv[(size_t)n * D + h * dh + t];
        outh[(size_t)m * D + h * dh + t] = (_Float16)(acc * inv);
    }
}

// gather token features (f16) and token xyz (*VOX)
__global__ void gather_tok(const _Float16* __restrict__ x3h, const float* __restrict__ coords3,
                           const int* __restrict__ inds, _Float16* __restrict__ tkh,
                           float* __restrict__ txyz, int C)
{
    int m = blockIdx.x, t = threadIdx.x;
    int src = inds[m];
    for (int c = t; c < C; c += blockDim.x)
        tkh[(size_t)m * C + c] = x3h[(size_t)src * C + c];
    if (t < 3) txyz[m * 3 + t] = coords3[src * 3 + t] * 0.05f;
}

// 3-NN (squared distance) + inverse-distance weights
__global__ void knn3(const float* __restrict__ coords3, const float* __restrict__ txyz,
                     int* __restrict__ idx3, float* __restrict__ w3, int Nn, int Mn)
{
    int n = blockIdx.x * blockDim.x + threadIdx.x;
    if (n >= Nn) return;
    float x = coords3[n * 3] * 0.05f, y = coords3[n * 3 + 1] * 0.05f, z = coords3[n * 3 + 2] * 0.05f;
    float d0 = 1e30f, d1 = 1e30f, d2 = 1e30f;
    int i0 = 0, i1 = 0, i2 = 0;
    for (int m = 0; m < Mn; ++m) {
        float dx = x - txyz[m * 3], dy = y - txyz[m * 3 + 1], dz = z - txyz[m * 3 + 2];
        float d = dx * dx + dy * dy + dz * dz;
        if (d < d0)      { d2 = d1; i2 = i1; d1 = d0; i1 = i0; d0 = d; i0 = m; }
        else if (d < d1) { d2 = d1; i2 = i1; d1 = d; i1 = m; }
        else if (d < d2) { d2 = d; i2 = m; }
    }
    float w0 = 1.f / (d0 + 1e-8f), w1 = 1.f / (d1 + 1e-8f), w2 = 1.f / (d2 + 1e-8f);
    float s = 1.f / (w0 + w1 + w2);
    idx3[n * 3] = i0; idx3[n * 3 + 1] = i1; idx3[n * 3 + 2] = i2;
    w3[n * 3] = w0 * s; w3[n * 3 + 1] = w1 * s; w3[n * 3 + 2] = w2 * s;
}

// inverse-distance interpolation, write f16 directly for the fp GEMM
__global__ void interp_k(const float* __restrict__ tok, const int* __restrict__ idx3,
                         const float* __restrict__ w3, _Float16* __restrict__ outh, int Nn)
{
    size_t i = (size_t)blockIdx.x * blockDim.x + threadIdx.x;
    if (i >= (size_t)Nn * 256) return;
    int n = (int)(i / 256), c = (int)(i % 256);
    float v = w3[n * 3]     * tok[(size_t)idx3[n * 3]     * 256 + c]
            + w3[n * 3 + 1] * tok[(size_t)idx3[n * 3 + 1] * 256 + c]
            + w3[n * 3 + 2] * tok[(size_t)idx3[n * 3 + 2] * 256 + c];
    outh[i] = (_Float16)v;
}

// ---------------------------------------------------------------------------
// Host orchestration
// ---------------------------------------------------------------------------
extern "C" void kernel_launch(void* const* d_in, const int* in_sizes, int n_in,
                              void* d_out, int out_size, void* d_ws, size_t ws_size,
                              hipStream_t stream)
{
    (void)in_sizes; (void)n_in; (void)out_size; (void)ws_size;
    // --- input pointers (dict insertion order; params dict expands in order) ---
    const float* feats   = (const float*)d_in[0];
    const float* coords3 = (const float*)d_in[1];
    auto F = [&](int i) { return (const float*)d_in[i]; };
    auto I = [&](int i) { return (const int*)d_in[i]; };
    const float *emb_w=F(2), *emb_g=F(3), *emb_b=F(4);
    const float *d2_w=F(5), *d2_g=F(6), *d2_b=F(7);
    const float *r1a_w=F(8), *r1a_g=F(9), *r1a_b=F(10);
    const float *r1b_w=F(11), *r1b_g=F(12), *r1b_b=F(13);
    const float *d4_w=F(14), *d4_g=F(15), *d4_b=F(16);
    const float *r2a_w=F(17), *r2a_g=F(18), *r2a_b=F(19);
    const float *r2b_w=F(20), *r2b_g=F(21), *r2b_b=F(22);
    const float *d8_w=F(23), *d8_g=F(24), *d8_b=F(25);
    const float *r3a_w=F(26), *r3a_g=F(27), *r3a_b=F(28);
    const float *r3b_w=F(29), *r3b_g=F(30), *r3b_b=F(31);
    const float *pre_w=F(32), *pre_b=F(33), *pos_w=F(34);
    const float *ln1_g=F(35), *ln1_b=F(36);
    const float *wq=F(37), *wk=F(38), *wv=F(39), *wo=F(40);
    const float *ln2_g=F(41), *ln2_b=F(42);
    const float *f1_w=F(43), *f1_b=F(44), *f2_w=F(45), *f2_b=F(46);
    const float *fp_w=F(47), *fp_g=F(48), *fp_b=F(49);
    const float *u4_w=F(50), *u4_g=F(51), *u4_b=F(52);
    const float *u4ra_w=F(53), *u4ra_g=F(54), *u4ra_b=F(55);
    const float *u4rb_w=F(56), *u4rb_g=F(57), *u4rb_b=F(58);
    const float *u2_w=F(59), *u2_g=F(60), *u2_b=F(61);
    const float *u2ra_w=F(62), *u2ra_g=F(63), *u2ra_b=F(64);
    const float *u2rb_w=F(65), *u2rb_g=F(66), *u2rb_b=F(67);
    const float *out_w=F(68), *out_g=F(69), *out_b=F(70);
    const int *nbr_d2=I(71), *nbr_r1=I(72), *nbr_d4=I(73), *nbr_r2=I(74);
    const int *nbr_d8=I(75), *nbr_r3=I(76), *nbr_u4=I(77), *nbr_u2=I(78);
    const int *inds=I(79);

    // --- workspace bump allocator (256B aligned) ---
    char* wsp = (char*)d_ws;
    size_t off = 0;
    auto allocB = [&](size_t bytes) -> void* {
        void* p = wsp + off;
        off = (off + bytes + 255) & ~(size_t)255;
        return p;
    };
    auto allocF = [&](size_t n) { return (float*)allocB(n * sizeof(float)); };
    auto allocH = [&](size_t n) { return (_Float16*)allocB(n * sizeof(_Float16)); };
    auto allocI = [&](size_t n) { return (int*)allocB(n * sizeof(int)); };

    float* stats = allocF(1024);
    float* scsh  = allocF(2048);
    int*   iota  = allocI(N1C);
    float* raw   = allocF((size_t)N1C * 128);   // shared conv/GEMM raw output
    float* x1    = allocF((size_t)N1C * 128);
    float* x2    = allocF((size_t)N2C * 256);
    float* x3    = allocF((size_t)N3C * 512);
    float* y2    = allocF((size_t)N2C * 256);
    float* y1    = allocF((size_t)N1C * 128);
    _Float16* x0h = allocH((size_t)N0C * 64);
    _Float16* x1h = allocH((size_t)N1C * 128);
    _Float16* x2h = allocH((size_t)N2C * 256);
    _Float16* x3h = allocH((size_t)N3C * 512);
    _Float16* h16 = allocH((size_t)N1C * 128);  // shared intermediate (max size)
    _Float16* y2h = allocH((size_t)N2C * 256);
    _Float16* y1h = allocH((size_t)N1C * 128);
    // transformer scratch
    _Float16* tkh   = allocH((size_t)MC * 512);
    float* txyz  = allocF((size_t)MC * 3);
    float* posb  = allocF((size_t)MC * 256);
    float* tok   = allocF((size_t)MC * 256);
    _Float16* hL = allocH((size_t)MC * 256);
    float* qf    = allocF((size_t)MC * 256);
    float* kf    = allocF((size_t)MC * 256);
    float* vf    = allocF((size_t)MC * 256);
    _Float16* attnh = allocH((size_t)MC * 256);
    float* t1    = allocF((size_t)MC * 256);
    _Float16* h2h = allocH((size_t)MC * 256);
    _Float16* hidh = allocH((size_t)MC * 1024);
    float* tok2  = allocF((size_t)MC * 256);
    int*   idx3  = allocI((size_t)N3C * 3);
    float* w3    = allocF((size_t)N3C * 3);
    _Float16* interph = allocH((size_t)N3C * 256);

    // packed f16 weights (WMMA fragment order)
    auto allocW = [&](int Cin, int Cout, int Kk) {
        return allocH((size_t)Kk * Cin * Cout);
    };
    _Float16 *pw_d2 = allocW(64, 128, KKC);
    _Float16 *pw_r1a = allocW(128, 128, KKC), *pw_r1b = allocW(128, 128, KKC);
    _Float16 *pw_d4 = allocW(128, 256, KKC);
    _Float16 *pw_r2a = allocW(256, 256, KKC), *pw_r2b = allocW(256, 256, KKC);
    _Float16 *pw_d8 = allocW(256, 512, KKC);
    _Float16 *pw_r3a = allocW(512, 512, KKC), *pw_r3b = allocW(512, 512, KKC);
    _Float16 *pw_pre = allocW(512, 256, 1);
    _Float16 *pw_q = allocW(256, 256, 1), *pw_k = allocW(256, 256, 1);
    _Float16 *pw_v = allocW(256, 256, 1), *pw_o = allocW(256, 256, 1);
    _Float16 *pw_f1 = allocW(256, 1024, 1), *pw_f2 = allocW(1024, 256, 1);
    _Float16 *pw_fp = allocW(256, 512, 1);
    _Float16 *pw_u4 = allocW(512, 256, KKC);
    _Float16 *pw_u4ra = allocW(256, 256, KKC), *pw_u4rb = allocW(256, 256, KKC);
    _Float16 *pw_u2 = allocW(256, 128, KKC);
    _Float16 *pw_u2ra = allocW(128, 128, KKC), *pw_u2rb = allocW(128, 128, KKC);
    _Float16 *pw_out = allocW(128, 64, 1);

    // --- helper launchers ---
    auto pack = [&](const float* w, _Float16* wp, int Cin, int Cout, int Kk) {
        size_t tot = (size_t)Kk * Cin * Cout;
        pack_w<<<dim3((unsigned)((tot + 255) / 256)), 256, 0, stream>>>(w, wp, Cin, Cout, Kk);
    };
    auto gemm = [&](const _Float16* A, const int* nb, const _Float16* W, float* O,
                    int Nout, int Cin, int Cout, int Kk) {
        int wpb = Cout / 32; if (wpb > 4) wpb = 4;
        dim3 blk(32, wpb);
        dim3 grd(Nout / 16, Cout / (32 * wpb));
        sconv_wmma<<<grd, blk, 0, stream>>>(A, nb, W, O, Nout, Cin, Cout, Kk);
    };
    auto apply = [&](const float* rawp, const float* sc, const float* sh,
                     const float* r1, const float* r2, float* of, _Float16* oh,
                     int N, int C, int relu) {
        size_t tot = (size_t)N * C;
        apply_ew<<<dim3((unsigned)((tot + 255) / 256)), 256, 0, stream>>>(
            rawp, sc, sh, r1, r2, of, oh, N, C, relu);
    };
    auto bnrun = [&](const float* rawp, const float* g, const float* b, int N, int C,
                     const float* r1, const float* r2, float* of, _Float16* oh) {
        zero_k<<<1, 256, 0, stream>>>(stats, 2 * C);
        bn_stats<<<256, 256, 0, stream>>>(rawp, stats, N, C);
        bn_finalize<<<(C + 63) / 64, 64, 0, stream>>>(stats, g, b, scsh, N, C);
        apply(rawp, scsh, scsh + C, r1, r2, of, oh, N, C, 1);
    };

    // --- 0. one-time prep: iota index map + weight packing ---
    iota_k<<<N1C / 256, 256, 0, stream>>>(iota, N1C);
    pack(d2_w, pw_d2, 64, 128, KKC);
    pack(r1a_w, pw_r1a, 128, 128, KKC);  pack(r1b_w, pw_r1b, 128, 128, KKC);
    pack(d4_w, pw_d4, 128, 256, KKC);
    pack(r2a_w, pw_r2a, 256, 256, KKC);  pack(r2b_w, pw_r2b, 256, 256, KKC);
    pack(d8_w, pw_d8, 256, 512, KKC);
    pack(r3a_w, pw_r3a, 512, 512, KKC);  pack(r3b_w, pw_r3b, 512, 512, KKC);
    pack(pre_w, pw_pre, 512, 256, 1);
    pack(wq, pw_q, 256, 256, 1);  pack(wk, pw_k, 256, 256, 1);
    pack(wv, pw_v, 256, 256, 1);  pack(wo, pw_o, 256, 256, 1);
    pack(f1_w, pw_f1, 256, 1024, 1);  pack(f2_w, pw_f2, 1024, 256, 1);
    pack(fp_w, pw_fp, 256, 512, 1);
    pack(u4_w, pw_u4, 512, 256, KKC);
    pack(u4ra_w, pw_u4ra, 256, 256, KKC);  pack(u4rb_w, pw_u4rb, 256, 256, KKC);
    pack(u2_w, pw_u2, 256, 128, KKC);
    pack(u2ra_w, pw_u2ra, 128, 128, KKC);  pack(u2rb_w, pw_u2rb, 128, 128, KKC);
    pack(out_w, pw_out, 128, 64, 1);

    // --- 1. embedding: x0 = relu(bn(feats @ emb_w)) ---
    matmul3<<<(N0C * 64) / 256, 256, 0, stream>>>(feats, emb_w, raw, N0C, 64);
    bnrun(raw, emb_g, emb_b, N0C, 64, nullptr, nullptr, nullptr, x0h);

    // --- 2. encoder stage 1 (N1, 128) ---
    gemm(x0h, nbr_d2, pw_d2, raw, N1C, 64, 128, KKC);
    bnrun(raw, d2_g, d2_b, N1C, 128, nullptr, nullptr, x1, x1h);
    gemm(x1h, nbr_r1, pw_r1a, raw, N1C, 128, 128, KKC);
    bnrun(raw, r1a_g, r1a_b, N1C, 128, nullptr, nullptr, nullptr, h16);
    gemm(h16, nbr_r1, pw_r1b, raw, N1C, 128, 128, KKC);
    bnrun(raw, r1b_g, r1b_b, N1C, 128, x1, nullptr, x1, x1h);

    // --- 3. encoder stage 2 (N2, 256) ---
    gemm(x1h, nbr_d4, pw_d4, raw, N2C, 128, 256, KKC);
    bnrun(raw, d4_g, d4_b, N2C, 256, nullptr, nullptr, x2, x2h);
    gemm(x2h, nbr_r2, pw_r2a, raw, N2C, 256, 256, KKC);
    bnrun(raw, r2a_g, r2a_b, N2C, 256, nullptr, nullptr, nullptr, h16);
    gemm(h16, nbr_r2, pw_r2b, raw, N2C, 256, 256, KKC);
    bnrun(raw, r2b_g, r2b_b, N2C, 256, x2, nullptr, x2, x2h);

    // --- 4. encoder stage 3 (N3, 512) ---
    gemm(x2h, nbr_d8, pw_d8, raw, N3C, 256, 512, KKC);
    bnrun(raw, d8_g, d8_b, N3C, 512, nullptr, nullptr, x3, x3h);
    gemm(x3h, nbr_r3, pw_r3a, raw, N3C, 512, 512, KKC);
    bnrun(raw, r3a_g, r3a_b, N3C, 512, nullptr, nullptr, nullptr, h16);
    gemm(h16, nbr_r3, pw_r3b, raw, N3C, 512, 512, KKC);
    bnrun(raw, r3b_g, r3b_b, N3C, 512, x3, nullptr, x3, x3h);

    // --- 5. tokens: tok = x3[inds] @ pre_w + pre_b + xyz @ pos_w ---
    gather_tok<<<MC, 256, 0, stream>>>(x3h, coords3, inds, tkh, txyz, 512);
    gemm(tkh, iota, pw_pre, raw, MC, 512, 256, 1);
    matmul3<<<(MC * 256) / 256, 256, 0, stream>>>(txyz, pos_w, posb, MC, 256);
    apply(raw, nullptr, pre_b, posb, nullptr, tok, nullptr, MC, 256, 0);

    // --- 6. transformer block ---
    ln_rows<<<MC, 256, 0, stream>>>(tok, ln1_g, ln1_b, hL, 256);
    gemm(hL, iota, pw_q, qf, MC, 256, 256, 1);
    gemm(hL, iota, pw_k, kf, MC, 256, 256, 1);
    gemm(hL, iota, pw_v, vf, MC, 256, 256, 1);
    attn_kernel<<<MC * 8, 128, 0, stream>>>(qf, kf, vf, attnh, MC);
    gemm(attnh, iota, pw_o, raw, MC, 256, 256, 1);
    apply(raw, nullptr, nullptr, tok, nullptr, t1, nullptr, MC, 256, 0);
    ln_rows<<<MC, 256, 0, stream>>>(t1, ln2_g, ln2_b, h2h, 256);
    gemm(h2h, iota, pw_f1, raw, MC, 256, 1024, 1);
    apply(raw, nullptr, f1_b, nullptr, nullptr, nullptr, hidh, MC, 1024, 1);
    gemm(hidh, iota, pw_f2, raw, MC, 1024, 256, 1);
    apply(raw, nullptr, f2_b, t1, nullptr, tok2, nullptr, MC, 256, 0);

    // --- 7. feature propagation: 3-NN interp + MLP, y = relu(bn(interp@fp_w)) + x3 ---
    knn3<<<(N3C + 63) / 64, 64, 0, stream>>>(coords3, txyz, idx3, w3, N3C, MC);
    interp_k<<<(N3C * 256) / 256, 256, 0, stream>>>(tok2, idx3, w3, interph, N3C);
    gemm(interph, iota, pw_fp, raw, N3C, 256, 512, 1);
    bnrun(raw, fp_g, fp_b, N3C, 512, nullptr, x3, x3, x3h);  // relu(bn)+x3, in place

    // --- 8. decoder stage 2 (N2, 256) ---
    gemm(x3h, nbr_u4, pw_u4, raw, N2C, 512, 256, KKC);
    bnrun(raw, u4_g, u4_b, N2C, 256, nullptr, nullptr, y2, y2h);
    gemm(y2h, nbr_r2, pw_u4ra, raw, N2C, 256, 256, KKC);
    bnrun(raw, u4ra_g, u4ra_b, N2C, 256, nullptr, nullptr, nullptr, h16);
    gemm(h16, nbr_r2, pw_u4rb, raw, N2C, 256, 256, KKC);
    bnrun(raw, u4rb_g, u4rb_b, N2C, 256, y2, x2, y2, y2h);   // relu(bn+y2)+x2

    // --- 9. decoder stage 1 (N1, 128) ---
    gemm(y2h, nbr_u2, pw_u2, raw, N1C, 256, 128, KKC);
    bnrun(raw, u2_g, u2_b, N1C, 128, nullptr, nullptr, y1, y1h);
    gemm(y1h, nbr_r1, pw_u2ra, raw, N1C, 128, 128, KKC);
    bnrun(raw, u2ra_g, u2ra_b, N1C, 128, nullptr, nullptr, nullptr, h16);
    gemm(h16, nbr_r1, pw_u2rb, raw, N1C, 128, 128, KKC);
    bnrun(raw, u2rb_g, u2rb_b, N1C, 128, y1, x1, y1, y1h);   // relu(bn+y1)+x1

    // --- 10. head: relu(bn(y1 @ out_w)) -> d_out ---
    gemm(y1h, iota, pw_out, raw, N1C, 128, 64, 1);
    bnrun(raw, out_g, out_b, N1C, 64, nullptr, nullptr, (float*)d_out, nullptr);
}
